// BNHCEncoder_77953656422746
// MI455X (gfx1250) — matrix operand order repaired and checked by reading, not compiled, not run on hardware
//
#include <hip/hip_runtime.h>
#include <stdint.h>

// ---------------------------------------------------------------------------
// MI455X (gfx1250) hypergraph conv encoder.
//   h = relu(x @ fc_w + fc_b)
//   repeat 3x:  xt = h @ W_l + b_l
//               hyper = segsum_col(xt[row] * coefA)     (E x 128)
//               z     = segsum_row(hyper[col] * coefB)  (N x 128)
//               h = relu(h + z)
// Dense GEMMs: v_wmma_f32_16x16x32_bf16, one 16x16 tile per wave32, A-tile
// staged to LDS by the Tensor Data Mover (tensor_load_to_lds + padded rows).
// Sparse phases: wave-per-edge float4 gather + global_atomic_add_f32 into
// L2-resident accumulators (xt/hyper_feat/z are 25.6 MB each << 192 MB L2).
// ---------------------------------------------------------------------------

#define DIMD   128          // feature dim D
#define LDSPAD 4            // LDS row pad (DWORDs) -> conflict-free frag reads

typedef __attribute__((ext_vector_type(16))) __bf16    v16bf;
typedef __attribute__((ext_vector_type(8)))  float     v8f;
typedef __attribute__((ext_vector_type(4)))  uint32_t  u32x4;
typedef __attribute__((ext_vector_type(8)))  int       i32x8;
typedef __attribute__((ext_vector_type(4)))  int       i32x4;

#if defined(__has_builtin)
#if __has_builtin(__builtin_amdgcn_tensor_load_to_lds)
#define USE_TDM 1
#endif
#endif

// -------------------------------- utility ----------------------------------

__global__ void zero_f32(float* __restrict__ p, int n) {
    int i = blockIdx.x * blockDim.x + threadIdx.x;
    if (i < n) p[i] = 0.0f;
}

// node_deg[row[m]] += weightMatrix[col[m]] ; hyper_deg[col[m]] += edgesWeights[m]
__global__ void accum_deg(const int* __restrict__ row, const int* __restrict__ col,
                          const float* __restrict__ wMat, const float* __restrict__ eW,
                          float* __restrict__ node_deg, float* __restrict__ hyper_deg,
                          int M) {
    int m = blockIdx.x * blockDim.x + threadIdx.x;
    if (m >= M) return;
    atomicAdd(&node_deg[row[m]],  wMat[col[m]]);
    atomicAdd(&hyper_deg[col[m]], eW[m]);
}

// deg -> (deg + 1e-8)^-0.5
__global__ void inv_sqrt_kernel(float* __restrict__ p, int n) {
    int i = blockIdx.x * blockDim.x + threadIdx.x;
    if (i < n) p[i] = rsqrtf(p[i] + 1e-8f);
}

// coefA[m] = eW[m]*norm ; coefB[m] = wMat[col[m]]*eW[m]*norm ; norm = hd[col]*nd[row]
__global__ void coef_kernel(const int* __restrict__ row, const int* __restrict__ col,
                            const float* __restrict__ wMat, const float* __restrict__ eW,
                            const float* __restrict__ nd, const float* __restrict__ hd,
                            float* __restrict__ coefA, float* __restrict__ coefB, int M) {
    int m = blockIdx.x * blockDim.x + threadIdx.x;
    if (m >= M) return;
    int r = row[m], c = col[m];
    float nrm = hd[c] * nd[r];
    float ew  = eW[m];
    coefA[m] = ew * nrm;
    coefB[m] = wMat[c] * ew * nrm;
}

// ------------------------------ dense GEMM ---------------------------------
// C[Nrows x 128] = act(A[Nrows x K] @ B[K x 128] + bias), K in {128, 256}.
// Block = 256 threads = 8 waves; block owns 16 rows; wave w owns cols [16w,16w+16).
// A-tile staged via TDM (tensor_load_to_lds, HW row padding = LDSPAD dwords),
// fragments converted f32->bf16 at consumption; B fragments coalesced from
// global (B is 64-128 KB, WGP$/L2-resident).
template <int K, bool RELU>
__global__ __launch_bounds__(256)
void gemm_bias_act(const float* __restrict__ A, const float* __restrict__ B,
                   const float* __restrict__ bias, float* __restrict__ C,
                   int Nrows) {
    constexpr int ldsK = K + LDSPAD;
    __shared__ float sA[16 * ldsK];
    const int tid   = threadIdx.x;
    const int tileM = blockIdx.x;

#if USE_TDM
    // ---- Tensor Data Mover: DMA the 16 x K f32 tile into LDS with per-row
    // padding (pad_interval = one K-dword row, pad_amount = LDSPAD dwords).
    if (tid < 32) {
        uint64_t gaddr = (uint64_t)(uintptr_t)(A + (size_t)tileM * 16 * K);
        uint32_t ldsb  = (uint32_t)(uintptr_t)&sA[0];     // low 32b = LDS offset
        constexpr int PAD_IV = (K == 256) ? 7 : 6;        // 1<<iv 8B units / row
        u32x4 g0;
        g0[0] = 1u;                                       // count=1 (valid user D#)
        g0[1] = ldsb;                                     // lds_addr
        g0[2] = (uint32_t)gaddr;                          // global_addr[31:0]
        g0[3] = (uint32_t)((gaddr >> 32) & 0x1FFFFFFu) | (2u << 30); // addr hi, type=2
        i32x8 g1;
        g1[0] = (2 << 16) | (1 << 20) | (PAD_IV << 22) | (3 << 25);
                 // data_size=4B | pad_enable | pad_interval | pad_amount=4 dw
        g1[1] = (int)((K & 0xFFFFu) << 16);               // tensor_dim0[15:0]
        g1[2] = (int)(((unsigned)K >> 16) | (((unsigned)Nrows & 0xFFFFu) << 16));
        g1[3] = (int)((((unsigned)Nrows >> 16) & 0xFFFFu) | ((unsigned)K << 16));
                 // tensor_dim1 hi | tile_dim0 = K
        g1[4] = 16;                                       // tile_dim1=16, tile_dim2=0
        g1[5] = K;                                        // tensor_dim0_stride lo32
        g1[6] = 0;                                        // stride hi, dim1_stride lo
        g1[7] = 0;
        i32x4 gz4 = {0, 0, 0, 0};                         // groups 2/3 unused (2D)
        i32x8 gz8 = {0, 0, 0, 0, 0, 0, 0, 0};
        __builtin_amdgcn_tensor_load_to_lds(g0, g1, gz4, gz4, gz8, 0);
        __builtin_amdgcn_s_wait_tensorcnt(0);
    }
    __syncthreads();
#else
    // ---- fallback: cooperative coalesced staging
#pragma unroll
    for (int idx = tid; idx < 16 * K; idx += 256) {
        int r  = idx / K;
        int kk = idx - r * K;
        int gr = tileM * 16 + r;
        if (gr >= Nrows) gr = Nrows - 1;
        sA[r * ldsK + kk] = A[(size_t)gr * K + kk];
    }
    __syncthreads();
#endif

    const int wave = tid >> 5;        // 0..7 -> column tile
    const int lane = tid & 31;
    const int half = lane >> 4;       // 0 | 1
    const int r16  = lane & 15;
    const int colBase = wave * 16;

    v8f acc = {};
#pragma unroll
    for (int k0 = 0; k0 < K; k0 += 32) {
        v16bf a, b;
#pragma unroll
        for (int j = 0; j < 8; ++j) {
            // A 16x32 bf16 frag: lane half selects K+8; VGPRs 4..7 hold K+16.
            int ka = ((j & 3) * 2) + 8 * half + 16 * (j >> 2);
            a[2 * j + 0] = (__bf16)sA[r16 * ldsK + k0 + ka + 0];
            a[2 * j + 1] = (__bf16)sA[r16 * ldsK + k0 + ka + 1];
            // B 32x16 bf16 frag: same K order as A (lane half selects K+8; VGPRs 4..7 hold K+16).
            int kb = ((j & 3) * 2) + 8 * half + 16 * (j >> 2);
            b[2 * j + 0] = (__bf16)B[(size_t)(k0 + kb + 0) * DIMD + colBase + r16];
            b[2 * j + 1] = (__bf16)B[(size_t)(k0 + kb + 1) * DIMD + colBase + r16];
        }
        acc = __builtin_amdgcn_wmma_f32_16x16x32_bf16(
                  false, a, false, b, (short)0, acc, false, false);
    }

    // C/D layout: VGPR j -> row j + 8*half, col = lane&15 (within tile)
    const int col = colBase + r16;
    const float bv = bias[col];
    const int rowBase = tileM * 16 + 8 * half;
    if (rowBase + 7 < Nrows) {                       // tile-exact fast path
#pragma unroll
        for (int j = 0; j < 8; ++j) {
            float v = acc[j] + bv;
            if (RELU) v = fmaxf(v, 0.0f);
            C[(size_t)(rowBase + j) * DIMD + col] = v;
        }
    } else {
#pragma unroll
        for (int j = 0; j < 8; ++j) {
            if (rowBase + j < Nrows) {
                float v = acc[j] + bv;
                if (RELU) v = fmaxf(v, 0.0f);
                C[(size_t)(rowBase + j) * DIMD + col] = v;
            }
        }
    }
}

// --------------------------- sparse scatter phase --------------------------
// One wave per incidence m: dst[sidx[m], :] += src[gidx[m], :] * coef[m]
// 32 lanes x float4 = 128 features.
__global__ __launch_bounds__(256)
void scatter_gather(const float* __restrict__ src, const int* __restrict__ gidx,
                    const int* __restrict__ sidx, const float* __restrict__ coef,
                    float* __restrict__ dst, int M) {
    int m = blockIdx.x * 8 + (threadIdx.x >> 5);
    if (m >= M) return;
    int lane = threadIdx.x & 31;
    int g = gidx[m];
    int s = sidx[m];
    float a = coef[m];
    float4 v = ((const float4*)(src + (size_t)g * DIMD))[lane];
    float* dp = dst + (size_t)s * DIMD + lane * 4;
    atomicAdd(dp + 0, v.x * a);
    atomicAdd(dp + 1, v.y * a);
    atomicAdd(dp + 2, v.z * a);
    atomicAdd(dp + 3, v.w * a);
}

__global__ void add_relu4(float* __restrict__ h, const float* __restrict__ z, int n4) {
    int i = blockIdx.x * blockDim.x + threadIdx.x;
    if (i >= n4) return;
    float4 a = ((const float4*)h)[i];
    float4 b = ((const float4*)z)[i];
    a.x = fmaxf(a.x + b.x, 0.0f);
    a.y = fmaxf(a.y + b.y, 0.0f);
    a.z = fmaxf(a.z + b.z, 0.0f);
    a.w = fmaxf(a.w + b.w, 0.0f);
    ((float4*)h)[i] = a;
}

// --------------------------------- driver ----------------------------------

extern "C" void kernel_launch(void* const* d_in, const int* in_sizes, int n_in,
                              void* d_out, int out_size, void* d_ws, size_t ws_size,
                              hipStream_t stream) {
    const float* x      = (const float*)d_in[0];
    const int*   hidx   = (const int*)  d_in[1];   // [2, M]
    const float* wMat   = (const float*)d_in[2];   // [E]
    const float* eW     = (const float*)d_in[3];   // [M]
    const float* fc_w   = (const float*)d_in[4];   // [256, 128]
    const float* fc_b   = (const float*)d_in[5];   // [128]
    const float* conv_w = (const float*)d_in[6];   // [L, 128, 128]
    const float* conv_b = (const float*)d_in[7];   // [L, 128]
    float*       h      = (float*)d_out;           // [N, 128]

    const int IN_DIM = 256, D = DIMD;
    const int N = in_sizes[0] / IN_DIM;            // 50000
    const int E = in_sizes[2];                     // 50000
    const int M = in_sizes[3];                     // 640000
    const int L = in_sizes[7] / D;                 // 3

    const int* row = hidx;
    const int* col = hidx + M;

    float* ws         = (float*)d_ws;
    float* node_deg   = ws;                 ws += N;           // -> nd^-1/2
    float* hyper_deg  = ws;                 ws += E;           // -> hd^-1/2
    float* coefA      = ws;                 ws += M;
    float* coefB      = ws;                 ws += M;
    float* xt         = ws;                 ws += (size_t)N * D;
    float* hyper_feat = ws;                 ws += (size_t)E * D;
    float* z          = ws;                 // [N, D] (contiguous after hyper_feat)

    const int T = 256;

    // ---- normalization coefficients (once, reused by all layers) ----
    zero_f32<<<(N + E + T - 1) / T, T, 0, stream>>>(node_deg, N + E);
    accum_deg<<<(M + T - 1) / T, T, 0, stream>>>(row, col, wMat, eW,
                                                 node_deg, hyper_deg, M);
    inv_sqrt_kernel<<<(N + E + T - 1) / T, T, 0, stream>>>(node_deg, N + E);
    coef_kernel<<<(M + T - 1) / T, T, 0, stream>>>(row, col, wMat, eW,
                                                   node_deg, hyper_deg,
                                                   coefA, coefB, M);

    // ---- input projection: h = relu(x @ fc_w + fc_b) ----
    gemm_bias_act<256, true><<<(N + 15) / 16, 256, 0, stream>>>(x, fc_w, fc_b, h, N);

    // ---- hypergraph conv layers ----
    for (int l = 0; l < L; ++l) {
        gemm_bias_act<128, false><<<(N + 15) / 16, 256, 0, stream>>>(
            h, conv_w + (size_t)l * D * D, conv_b + (size_t)l * D, xt, N);

        // hyper_feat and z are contiguous: zero both in one launch
        zero_f32<<<(((size_t)(E + N) * D + T - 1) / T), T, 0, stream>>>(
            hyper_feat, (E + N) * D);

        // hyper_feat[col] += xt[row] * coefA
        scatter_gather<<<(M + 7) / 8, 256, 0, stream>>>(
            xt, row, col, coefA, hyper_feat, M);
        // z[row] += hyper_feat[col] * coefB
        scatter_gather<<<(M + 7) / 8, 256, 0, stream>>>(
            hyper_feat, col, row, coefB, z, M);

        add_relu4<<<((size_t)N * D / 4 + T - 1) / T, T, 0, stream>>>(h, z, N * D / 4);
    }
}
